// ClusterLocalAttention_77807627535045
// MI455X (gfx1250) — matrix operand order repaired and checked by reading, not compiled, not run on hardware
//
#include <hip/hip_runtime.h>

// ---------------- problem constants ----------------
constexpr int L  = 32768;
constexpr int C  = 384;
constexpr int H  = 4;
constexpr int D  = 96;          // C / H
constexpr int NC = 512;
constexpr int CS = 64;
constexpr int C3 = 3 * C;       // 1152
constexpr int MAXW = 2048;      // generous bound on number of windows (~1024 worst case)

// workspace int-offsets (meta region)
constexpr int OFF_HIST   = 0;        // 512
constexpr int OFF_CSTART = 512;      // 512
constexpr int OFF_WSTART = 1024;     // 2048
constexpr int OFF_WSIZE  = 3072;     // 2048
constexpr int OFF_NW     = 5120;     // 1
constexpr int OFF_IDX    = 5632;     // L
constexpr size_t QKV_OFF = 256 * 1024;                    // byte offset, past meta
constexpr size_t H_OFF   = QKV_OFF + (size_t)L * C3 * 2;  // f16 qkv then f16 h

// ---------------- types ----------------
typedef __attribute__((ext_vector_type(16))) _Float16 v16h;
typedef __attribute__((ext_vector_type(8)))  float    v8f;
typedef __attribute__((ext_vector_type(4)))  unsigned int u32x4;
typedef __attribute__((ext_vector_type(8)))  unsigned int u32x8;

__device__ inline v8f wmma16(v16h a, v16h b, v8f c) {
  return __builtin_amdgcn_wmma_f32_16x16x32_f16(false, a, false, b, (short)0, c, false, false);
}

// A fragment: 16x32 f16 row-major source, row stride rs (elements).
// lanes 0-15: M=lane, K halves {0..7,16..23}; lanes 16-31: M=lane-16, K {8..15,24..31}
__device__ inline v16h load_a_frag(const _Float16* base, int rs, int lane) {
  int m = lane & 15, hi = lane >> 4;
  v16h a;
#pragma unroll
  for (int v = 0; v < 8; ++v) {
    int k = (v & 3) * 2 + hi * 8 + ((v >= 4) ? 16 : 0);
    a[2 * v]     = base[m * rs + k];
    a[2 * v + 1] = base[m * rs + k + 1];
  }
  return a;
}

// B fragment: 32x16 f16, source row-major [k][n] with stride rs.
__device__ inline v16h load_b_frag(const _Float16* base, int rs, int lane) {
  int n = lane & 15, hi = lane >> 4;
  v16h b;
#pragma unroll
  for (int v = 0; v < 8; ++v) {
    int k = (v & 3) * 2 + hi * 8 + ((v >= 4) ? 16 : 0);
    b[2 * v]     = base[k * rs + n];
    b[2 * v + 1] = base[(k + 1) * rs + n];
  }
  return b;
}

// B fragment from a transposed source: element (k,n) = base[n*rs + k] (pairs contiguous)
__device__ inline v16h load_bT_frag(const _Float16* base, int rs, int lane) {
  int n = lane & 15, hi = lane >> 4;
  v16h b;
#pragma unroll
  for (int v = 0; v < 8; ++v) {
    int k = (v & 3) * 2 + hi * 8 + ((v >= 4) ? 16 : 0);
    b[2 * v]     = base[n * rs + k];
    b[2 * v + 1] = base[n * rs + k + 1];
  }
  return b;
}

// ---- TDM: 2D tile (rows x 96 f16, row stride C3 elems) global -> LDS, zero-fill OOB rows
// D# layout per CDNA5 ISA; 2-group form (<=2D tensor).
__device__ inline void tdm_load_tile_2d(unsigned lds_addr, unsigned long long gaddr,
                                        int valid_rows) {
  u32x4 g0;
  g0[0] = 1u;                                     // count=1, user mode
  g0[1] = lds_addr;                               // lds_addr
  g0[2] = (unsigned)(gaddr & 0xffffffffull);      // global_addr[31:0]
  g0[3] = (unsigned)((gaddr >> 32) & 0x1ffffffull) | (2u << 30);  // addr[56:32] | type=2
  u32x8 g1;
  g1[0] = (1u << 16);                             // workgroup_mask=0, data_size=1 (2B)
  g1[1] = (unsigned)(D & 0xffff) << 16;           // tensor_dim0[15:0] in bits 63:48
  g1[2] = ((unsigned)valid_rows & 0xffffu) << 16; // tensor_dim0 hi=0 | tensor_dim1[15:0]
  g1[3] = (((unsigned)valid_rows >> 16) & 0xffffu) | ((unsigned)D << 16); // dim1 hi | tile_dim0
  g1[4] = 128u;                                   // tile_dim1=128 (OOB rows -> zero), tile_dim2=0
  g1[5] = (unsigned)C3;                           // tensor_dim0_stride[31:0] = 1152
  g1[6] = 0u;                                     // stride0 hi | stride1 lo
  g1[7] = 0u;
  asm volatile("tensor_load_to_lds %0, %1" : : "s"(g0), "s"(g1) : "memory");
}

// ---------------- setup kernels ----------------
__global__ void zero_meta_kernel(int* meta) {
  int i = blockIdx.x * blockDim.x + threadIdx.x;
  if (i < OFF_IDX) meta[i] = 0;
}

__global__ void hist_kernel(const int* __restrict__ labels, int* meta) {
  int i = blockIdx.x * blockDim.x + threadIdx.x;
  atomicAdd(&meta[OFF_HIST + labels[i]], 1);
}

// one block of 512 threads: cluster starts + window decomposition
__global__ void plan_kernel(int* meta) {
  __shared__ int sscan[NC];
  __shared__ int wscan[NC];
  int t = threadIdx.x;
  int s = meta[OFF_HIST + t];
  sscan[t] = s;
  for (int off = 1; off < NC; off <<= 1) {
    __syncthreads();
    int v = (t >= off) ? sscan[t - off] : 0;
    __syncthreads();
    sscan[t] += v;
  }
  __syncthreads();
  int cst = sscan[t] - s;
  meta[OFF_CSTART + t] = cst;

  int n = (s >= 2 * CS) ? (s / CS) : (s > 0 ? 1 : 0);
  wscan[t] = n;
  for (int off = 1; off < NC; off <<= 1) {
    __syncthreads();
    int v = (t >= off) ? wscan[t - off] : 0;
    __syncthreads();
    wscan[t] += v;
  }
  __syncthreads();
  int wbase = wscan[t] - n;
  if (t == NC - 1) meta[OFF_NW] = wscan[t];
  if (n > 0) {
    int q = s / n, r = s % n, st = cst;
    for (int i = 0; i < n; ++i) {
      int sz = q + (i < r ? 1 : 0);
      meta[OFF_WSTART + wbase + i] = st;
      meta[OFF_WSIZE  + wbase + i] = sz;
      st += sz;
    }
  }
}

// stable counting-sort scatter: one wave per cluster, ballot-based ranks
__global__ void sort_kernel(const int* __restrict__ labels, int* meta) {
  int c = blockIdx.x;
  int lane = threadIdx.x;                 // 32 threads
  int base = meta[OFF_CSTART + c];
  int cnt = 0;
  for (int i0 = 0; i0 < L; i0 += 32) {
    int i = i0 + lane;
    bool mch = (labels[i] == c);
    unsigned long long bal = __ballot(mch);
    if (mch) {
      int pos = cnt + __popcll(bal & ((1ull << lane) - 1ull));
      meta[OFF_IDX + base + pos] = i;
    }
    cnt += __popcll(bal);
  }
}

// ---------------- QKV projection: qkv[l,:] = f16( x[index[l],:] @ w_qkv + b ) ----------------
__global__ __launch_bounds__(256) void qkv_gemm_kernel(const float* __restrict__ x,
                                                       const float* __restrict__ w,
                                                       const float* __restrict__ bias,
                                                       const int* __restrict__ idx,
                                                       _Float16* __restrict__ qkv) {
  __shared__ _Float16 As[128 * 32];
  __shared__ _Float16 Bs[32 * 64];
  int row0 = blockIdx.x * 128;
  int n0   = blockIdx.y * 64;
  int tid = threadIdx.x, lane = tid & 31, wave = tid >> 5;
  v8f acc[4] = {};
  for (int k0 = 0; k0 < C; k0 += 32) {
    {
      int r = tid >> 1, kk0 = (tid & 1) * 16;
      int grow = idx[row0 + r];
      const float* src = x + (size_t)grow * C + k0 + kk0;
      __builtin_prefetch(src + 32, 0, 1);
#pragma unroll
      for (int j = 0; j < 16; ++j) As[r * 32 + kk0 + j] = (_Float16)src[j];
    }
    {
      int kr = tid >> 3, c0 = (tid & 7) * 8;
      const float* src = w + (size_t)(k0 + kr) * C3 + n0 + c0;
#pragma unroll
      for (int j = 0; j < 8; ++j) Bs[kr * 64 + c0 + j] = (_Float16)src[j];
    }
    __syncthreads();
    v16h a = load_a_frag(As + wave * 16 * 32, 32, lane);
#pragma unroll
    for (int t = 0; t < 4; ++t) {
      v16h b = load_b_frag(Bs + t * 16, 64, lane);
      acc[t] = wmma16(a, b, acc[t]);
    }
    __syncthreads();
  }
  int m = lane & 15, hi = lane >> 4;
#pragma unroll
  for (int t = 0; t < 4; ++t) {
    int col = n0 + t * 16 + m;
    float bv = bias[col];
#pragma unroll
    for (int v = 0; v < 8; ++v) {
      int row = row0 + wave * 16 + v + 8 * hi;
      qkv[(size_t)row * C3 + col] = (_Float16)(acc[t][v] + bv);
    }
  }
}

// ---------------- per-(window, head) attention ----------------
__global__ __launch_bounds__(256) void attn_kernel(const _Float16* __restrict__ qkv,
                                                   _Float16* __restrict__ hbuf,
                                                   const int* __restrict__ meta) {
  int wid = blockIdx.x;
  if (wid >= meta[OFF_NW]) return;
  int head  = blockIdx.y;
  int start = meta[OFF_WSTART + wid];
  int s     = meta[OFF_WSIZE + wid];
  int tid = threadIdx.x, lane = tid & 31, wave = tid >> 5;

  // 56 KB static LDS: V (128x96 f16) + union{ K 128x96 f16 , P 128x128 f16 }
  __shared__ _Float16 smem[128 * 96 + 128 * 128];
  _Float16* Vs = smem;
  _Float16* Ks = smem + 128 * 96;
  _Float16* Ps = smem + 128 * 96;

  // K and V tiles via Tensor Data Mover: rows >= s come back zero (OOB of tensor_dim1)
  if (wave == 0) {
    unsigned long long base = (unsigned long long)(uintptr_t)(qkv + (size_t)start * C3);
    tdm_load_tile_2d((unsigned)(uintptr_t)Ks, base + (size_t)(C     + head * D) * 2, s);
    tdm_load_tile_2d((unsigned)(uintptr_t)Vs, base + (size_t)(2 * C + head * D) * 2, s);
    __builtin_amdgcn_s_wait_tensorcnt(0);
  }
  __syncthreads();

  int r0 = wave * 16;
  int m = lane & 15, hi = lane >> 4;

  // scores: S = Q @ K^T ; Q fragments straight from global (L2-resident)
  v8f st[8] = {};
#pragma unroll
  for (int kc = 0; kc < 3; ++kc) {
    v16h a;
#pragma unroll
    for (int v = 0; v < 8; ++v) {
      int k = (v & 3) * 2 + hi * 8 + ((v >= 4) ? 16 : 0) + kc * 32;
      int row = start + r0 + m;
      if (row >= L) row = L - 1;
      const _Float16* src = qkv + (size_t)row * C3 + head * D + k;
      a[2 * v] = src[0];
      a[2 * v + 1] = src[1];
    }
#pragma unroll
    for (int t = 0; t < 8; ++t) {
      v16h b = load_bT_frag(Ks + (t * 16) * 96 + kc * 32, 96, lane);
      st[t] = wmma16(a, b, st[t]);
    }
  }

  // scale + key-mask + row softmax (register, 16-lane reductions)
  const float scale = 0.1020620726f;   // 1/sqrt(96)
#pragma unroll
  for (int t = 0; t < 8; ++t) {
    int col = t * 16 + m;
#pragma unroll
    for (int v = 0; v < 8; ++v) {
      float val = st[t][v] * scale;
      if (col >= s) val = -1.0e9f;
      st[t][v] = val;
    }
  }
#pragma unroll
  for (int v = 0; v < 8; ++v) {
    float mx = -3.0e38f;
#pragma unroll
    for (int t = 0; t < 8; ++t) mx = fmaxf(mx, st[t][v]);
#pragma unroll
    for (int off = 1; off < 16; off <<= 1) mx = fmaxf(mx, __shfl_xor(mx, off, 32));
    float sum = 0.0f;
#pragma unroll
    for (int t = 0; t < 8; ++t) { float p = __expf(st[t][v] - mx); st[t][v] = p; sum += p; }
#pragma unroll
    for (int off = 1; off < 16; off <<= 1) sum += __shfl_xor(sum, off, 32);
    float inv = 1.0f / sum;
#pragma unroll
    for (int t = 0; t < 8; ++t) st[t][v] *= inv;
  }
  __syncthreads();                 // everyone done reading K

  // write probabilities (f16) into LDS, overlaying K region
#pragma unroll
  for (int t = 0; t < 8; ++t) {
    int col = t * 16 + m;
#pragma unroll
    for (int v = 0; v < 8; ++v) {
      int row = r0 + v + 8 * hi;
      Ps[row * 128 + col] = (_Float16)st[t][v];
    }
  }
  __syncthreads();

  // O = P @ V
  v8f oc[6] = {};
#pragma unroll
  for (int kc = 0; kc < 4; ++kc) {
    v16h a = load_a_frag(Ps + r0 * 128 + kc * 32, 128, lane);
#pragma unroll
    for (int t = 0; t < 6; ++t) {
      v16h b = load_b_frag(Vs + (kc * 32) * 96 + t * 16, 96, lane);
      oc[t] = wmma16(a, b, oc[t]);
    }
  }
#pragma unroll
  for (int t = 0; t < 6; ++t) {
    int d = head * D + t * 16 + m;
#pragma unroll
    for (int v = 0; v < 8; ++v) {
      int row = r0 + v + 8 * hi;
      if (row < s) hbuf[(size_t)(start + row) * C + d] = (_Float16)oc[t][v];
    }
  }
}

// ---------------- output projection: out = h @ w_out + b_out + x[index] ----------------
__global__ __launch_bounds__(256) void out_gemm_kernel(const _Float16* __restrict__ hbuf,
                                                       const float* __restrict__ w,
                                                       const float* __restrict__ bias,
                                                       const float* __restrict__ x,
                                                       const int* __restrict__ idx,
                                                       float* __restrict__ out) {
  __shared__ _Float16 As[128 * 32];
  __shared__ _Float16 Bs[32 * 64];
  int row0 = blockIdx.x * 128;
  int n0   = blockIdx.y * 64;
  int tid = threadIdx.x, lane = tid & 31, wave = tid >> 5;
  v8f acc[4] = {};
  for (int k0 = 0; k0 < C; k0 += 32) {
    {
      // pure f16 copy -> async global->LDS (2 x B128 per thread), ASYNCcnt-tracked
      int r = tid >> 1, kk0 = (tid & 1) * 16;
      unsigned long long ga =
          (unsigned long long)(uintptr_t)(hbuf + (size_t)(row0 + r) * C + k0 + kk0);
      unsigned ldsa = (unsigned)(uintptr_t)(As + r * 32 + kk0);
      asm volatile("global_load_async_to_lds_b128 %0, %1, off\n\t"
                   "global_load_async_to_lds_b128 %0, %1, off offset:16"
                   : : "v"(ldsa), "v"(ga) : "memory");
    }
    {
      int kr = tid >> 3, c0 = (tid & 7) * 8;
      const float* src = w + (size_t)(k0 + kr) * C + n0 + c0;
#pragma unroll
      for (int j = 0; j < 8; ++j) Bs[kr * 64 + c0 + j] = (_Float16)src[j];
    }
    asm volatile("s_wait_asynccnt 0x0" ::: "memory");
    __syncthreads();
    v16h a = load_a_frag(As + wave * 16 * 32, 32, lane);
#pragma unroll
    for (int t = 0; t < 4; ++t) {
      v16h b = load_b_frag(Bs + t * 16, 64, lane);
      acc[t] = wmma16(a, b, acc[t]);
    }
    __syncthreads();
  }
  int m = lane & 15, hi = lane >> 4;
#pragma unroll
  for (int t = 0; t < 4; ++t) {
    int col = n0 + t * 16 + m;
    float bv = bias[col];
#pragma unroll
    for (int v = 0; v < 8; ++v) {
      int row = row0 + wave * 16 + v + 8 * hi;
      float xs = x[(size_t)idx[row] * C + col];
      out[(size_t)row * C + col] = acc[t][v] + bv + xs;
    }
  }
}

// ---------------- launcher ----------------
extern "C" void kernel_launch(void* const* d_in, const int* in_sizes, int n_in,
                              void* d_out, int out_size, void* d_ws, size_t ws_size,
                              hipStream_t stream) {
  const float* x      = (const float*)d_in[0];
  const int*   labels = (const int*)d_in[1];
  const float* w_qkv  = (const float*)d_in[2];
  const float* b_qkv  = (const float*)d_in[3];
  const float* w_out  = (const float*)d_in[4];
  const float* b_out  = (const float*)d_in[5];
  float* out = (float*)d_out;

  int* meta = (int*)d_ws;
  char* wsb = (char*)d_ws;
  _Float16* qkv  = (_Float16*)(wsb + QKV_OFF);
  _Float16* hbuf = (_Float16*)(wsb + H_OFF);
  const int* idx = meta + OFF_IDX;

  zero_meta_kernel<<<(OFF_IDX + 255) / 256, 256, 0, stream>>>(meta);
  hist_kernel<<<L / 256, 256, 0, stream>>>(labels, meta);
  plan_kernel<<<1, NC, 0, stream>>>(meta);
  sort_kernel<<<NC, 32, 0, stream>>>(labels, meta);
  qkv_gemm_kernel<<<dim3(L / 128, C3 / 64), 256, 0, stream>>>(x, w_qkv, b_qkv, idx, qkv);
  attn_kernel<<<dim3(MAXW, H), 256, 0, stream>>>(qkv, hbuf, meta);
  out_gemm_kernel<<<dim3(L / 128, C / 64), 256, 0, stream>>>(hbuf, w_out, b_out, x, idx, out);
}
